// LSTMMeasurementSelector_36713380446679
// MI455X (gfx1250) — compile-verified, matrix operand order checked by reading, not run on hardware
//
#include <hip/hip_runtime.h>

typedef __attribute__((ext_vector_type(16))) _Float16 v16h;
typedef __attribute__((ext_vector_type(8)))  float    v8f;

static constexpr int NQ = 9;
static constexpr int KC = 3;
static constexpr int SS = 16;   // S steps
static constexpr int HH = 128;
static constexpr int BB = 16;
static constexpr int DD = 512;  // 2^NQ
static constexpr int XDIM = NQ + NQ * KC;     // 36
static constexpr int KU = 192;                // padded concat: 64 (x) + 128 (h)
static constexpr int GN = 4 * HH;             // 512 gates

// ---- workspace layout (bytes) ----
static constexpr size_t WCAT_OFF  = 0;                          // f16 [512][192]
static constexpr size_t BIAS_OFF  = WCAT_OFF  + 512*KU*2;       // f32 [512]
static constexpr size_t BPROJ_OFF = BIAS_OFF  + 512*4;          // f16 [128][32]
static constexpr size_t BPPAD_OFF = BPROJ_OFF + HH*32*2;        // f32 [32]
static constexpr size_t INPUTS_OFF= BPPAD_OFF + 32*4;           // f32 [S][B][36]
static constexpr size_t RHOQ_OFF  = INPUTS_OFF+ (size_t)SS*BB*XDIM*4; // f32 [B][S][NQ][8]
static constexpr size_t WS_NEED   = RHOQ_OFF  + (size_t)BB*SS*NQ*8*4;

// ---------------- k_init: pack weights into WMMA-friendly f16 buffers -------------
__global__ void k_init(const float* __restrict__ W_ih, const float* __restrict__ W_hh,
                       const float* __restrict__ b_ih, const float* __restrict__ b_hh,
                       const float* __restrict__ Wp,   const float* __restrict__ bp,
                       _Float16* __restrict__ Wcat, float* __restrict__ biasSum,
                       _Float16* __restrict__ Bproj, float* __restrict__ bpPad) {
  int t = blockIdx.x * blockDim.x + threadIdx.x;
  if (t < 512 * KU) {
    int n = t / KU, k = t % KU;
    float v = 0.f;
    if (k < XDIM)      v = W_ih[n * XDIM + k];
    else if (k >= 64)  v = W_hh[n * HH + (k - 64)];
    Wcat[t] = (_Float16)v;
  }
  if (t < 512) biasSum[t] = b_ih[t] + b_hh[t];
  if (t < HH * 32) {
    int kk = t / 32, n = t % 32;      // Bproj[k][n] = Wp_flat[n*128 + k], n<27
    Bproj[t] = (_Float16)((n < NQ * KC) ? Wp[n * HH + kk] : 0.f);
  }
  if (t < 32) bpPad[t] = (t < NQ * KC) ? bp[t] : 0.f;
}

// ---------------- k_phase1: persistent single-WGP recurrent kernel ----------------
__global__ void __launch_bounds__(256) k_phase1(
    const float* __restrict__ snapshot, const float* __restrict__ bcv,
    const float* __restrict__ rho, const float* __restrict__ h0, const float* __restrict__ c0,
    const _Float16* __restrict__ Wcat, const float* __restrict__ biasSum,
    const _Float16* __restrict__ Bproj, const float* __restrict__ bpPad,
    float* __restrict__ inputs, float* __restrict__ lastbv_out) {
  extern __shared__ char smem[];
  _Float16* uf16 = (_Float16*)smem;              // [16][192]
  _Float16* hf16 = uf16 + BB * KU;               // [16][128]
  float* g    = (float*)(hf16 + BB * HH);        // [16][512]
  float* hS   = g    + BB * GN;                  // [16][128]
  float* cS   = hS   + BB * HH;                  // [16][128]
  float* psiR = cS   + BB * HH;                  // [16][512]
  float* psiI = psiR + BB * DD;                  // [16][512]
  float* xcur = psiI + BB * DD;                  // [16][36]
  float* amps = xcur + BB * XDIM;                // [16][27]
  float* logits = amps + BB * NQ * KC;           // [16][32]
  float* probP  = logits + BB * 32;              // [16]
  float* probM  = probP + BB;                    // [16]
  float* scaleS = probM + BB;                    // [16]

  const int tid = threadIdx.x;
  const int lane = tid & 31;
  const int wid  = tid >> 5;
  const int laneM = lane & 15;
  const int halfSel = lane >> 4;

  // ---- init state ----
  for (int i = tid; i < BB * HH; i += 256) { hS[i] = h0[i]; cS[i] = c0[i]; }
  for (int i = tid; i < BB * DD; i += 256) {
    int b = i / DD, idx = i % DD;
    psiR[i] = rho[(b * 2 + 0) * DD + idx];
    psiI[i] = rho[(b * 2 + 1) * DD + idx];
  }
  for (int i = tid; i < BB * XDIM; i += 256) {
    int b = i / XDIM, j = i % XDIM;
    float v = (j < NQ) ? snapshot[b * NQ + j] : bcv[b * NQ * KC + (j - NQ)];
    xcur[i] = v;
    inputs[i] = v;                               // inputs[0]
  }
  __syncthreads();

  for (int s = 1; s < SS; ++s) {
    // ---- build u = [x | 0pad | h] in f16 ----
    for (int i = tid; i < BB * KU; i += 256) {
      int b = i / KU, k = i % KU;
      float v = (k < XDIM) ? xcur[b * XDIM + k] : (k < 64 ? 0.f : hS[b * HH + (k - 64)]);
      uf16[i] = (_Float16)v;
    }
    __syncthreads();

    // ---- GEMM: g[16][512] = u[16][192] @ Wcat^T + bias  (WMMA 16x16x32 f16) ----
    for (int t = wid; t < GN / 16; t += 8) {
      int n0 = t * 16;
      __builtin_prefetch(&Wcat[(n0 + laneM) * KU], 0, 0);
      v8f acc;
      float bias = biasSum[n0 + laneM];
      for (int r = 0; r < 8; ++r) acc[r] = bias;
      for (int kk = 0; kk < KU / 32; ++kk) {
        v16h a, bf;
        for (int e = 0; e < 16; ++e) {
          int K = kk * 32 + ((e < 8) ? 0 : 16) + halfSel * 8 + (e & 7);
          a[e]  = uf16[laneM * KU + K];
          bf[e] = Wcat[(n0 + laneM) * KU + K];   // B[K][N] = Wcat[N][K]
        }
        acc = __builtin_amdgcn_wmma_f32_16x16x32_f16(false, a, false, bf,
                                                     (short)0, acc, false, false);
      }
      for (int r = 0; r < 8; ++r)
        g[(r + halfSel * 8) * GN + n0 + laneM] = acc[r];
    }
    __syncthreads();

    // ---- LSTM gate update ----
    for (int i = tid; i < BB * HH; i += 256) {
      int b = i / HH, j = i % HH;
      float gi = g[b * GN + j];
      float gf = g[b * GN + HH + j];
      float gg = g[b * GN + 2 * HH + j];
      float go = g[b * GN + 3 * HH + j];
      float ii = 1.f / (1.f + __expf(-gi));
      float ff = 1.f / (1.f + __expf(-gf));
      float oo = 1.f / (1.f + __expf(-go));
      float cn = ff * cS[i] + ii * tanhf(gg);
      cS[i] = cn;
      float hn = oo * tanhf(cn);
      hS[i] = hn;
      hf16[i] = (_Float16)hn;
    }
    __syncthreads();

    // ---- projection logits[16][32] = h @ Bproj + bp  (2 WMMA tiles) ----
    if (wid < 2) {
      int n0 = wid * 16;
      v8f acc;
      float bias = bpPad[n0 + laneM];
      for (int r = 0; r < 8; ++r) acc[r] = bias;
      for (int kk = 0; kk < HH / 32; ++kk) {
        v16h a, bf;
        for (int e = 0; e < 16; ++e) {
          int K = kk * 32 + ((e < 8) ? 0 : 16) + halfSel * 8 + (e & 7);
          a[e]  = hf16[laneM * HH + K];
          bf[e] = Bproj[K * 32 + n0 + laneM];
        }
        acc = __builtin_amdgcn_wmma_f32_16x16x32_f16(false, a, false, bf,
                                                     (short)0, acc, false, false);
      }
      for (int r = 0; r < 8; ++r)
        logits[(r + halfSel * 8) * 32 + n0 + laneM] = acc[r];
    }
    __syncthreads();

    // ---- softmax over K=3, amps = sqrt(probs) ----
    if (tid < BB * NQ) {
      int b = tid / NQ, q = tid % NQ;
      float l0 = logits[b * 32 + q * 3 + 0];
      float l1 = logits[b * 32 + q * 3 + 1];
      float l2 = logits[b * 32 + q * 3 + 2];
      float m = fmaxf(l0, fmaxf(l1, l2));
      float e0 = __expf(l0 - m), e1 = __expf(l1 - m), e2 = __expf(l2 - m);
      float inv = 1.f / (e0 + e1 + e2);
      float a0 = sqrtf(e0 * inv), a1 = sqrtf(e1 * inv), a2 = sqrtf(e2 * inv);
      amps[(b * NQ + q) * 3 + 0] = a0;
      amps[(b * NQ + q) * 3 + 1] = a1;
      amps[(b * NQ + q) * 3 + 2] = a2;
      xcur[b * XDIM + NQ + q * 3 + 0] = a0;
      xcur[b * XDIM + NQ + q * 3 + 1] = a1;
      xcur[b * XDIM + NQ + q * 3 + 2] = a2;
    }
    if (tid < BB) { probP[tid] = 0.f; probM[tid] = 0.f; }
    __syncthreads();

    // ---- snapshot collapse: 9 sequential per-qubit 2x2 projector applications ----
    for (int q = 0; q < NQ; ++q) {
      int stride = DD >> (q + 1);                // 256 >> q
      for (int b = 0; b < BB; ++b) {
        float a1 = amps[(b * NQ + q) * 3 + 0];
        float a2 = amps[(b * NQ + q) * 3 + 1];
        float a3 = amps[(b * NQ + q) * 3 + 2];
        float an = sqrtf(a1 * a1 + a2 * a2 + a3 * a3);
        float n2 = 1.f / (2.f * an * (a3 + an));  // nrm^2
        float t0 = a3 + an;
        float Pp00 = t0 * t0 * n2;
        float Pp01r = t0 * a1 * n2, Pp01i = -t0 * a2 * n2;
        float Pp11 = (a1 * a1 + a2 * a2) * n2;
        float Pm00 = Pp11;
        float Pm01r = -a1 * t0 * n2, Pm01i = a2 * t0 * n2;
        float Pm11 = Pp00;

        int p = tid;                             // one pair per thread, 256 pairs
        int low = p & (stride - 1);
        int high = p >> (8 - q);
        int i0 = b * DD + high * 2 * stride + low;
        int i1 = i0 + stride;
        float v0r = psiR[i0], v0i = psiI[i0];
        float v1r = psiR[i1], v1i = psiI[i1];
        // plus branch (P10 = conj(P01))
        float p0r = Pp00 * v0r + Pp01r * v1r - Pp01i * v1i;
        float p0i = Pp00 * v0i + Pp01r * v1i + Pp01i * v1r;
        float p1r = Pp01r * v0r + Pp01i * v0i + Pp11 * v1r;
        float p1i = Pp01r * v0i - Pp01i * v0r + Pp11 * v1i;
        // minus branch (probability only)
        float m0r = Pm00 * v0r + Pm01r * v1r - Pm01i * v1i;
        float m0i = Pm00 * v0i + Pm01r * v1i + Pm01i * v1r;
        float m1r = Pm01r * v0r + Pm01i * v0i + Pm11 * v1r;
        float m1i = Pm01r * v0i - Pm01i * v0r + Pm11 * v1i;
        psiR[i0] = p0r; psiI[i0] = p0i;
        psiR[i1] = p1r; psiI[i1] = p1i;
        atomicAdd(&probP[b], p0r * p0r + p0i * p0i + p1r * p1r + p1i * p1i);
        atomicAdd(&probM[b], m0r * m0r + m0i * m0i + m1r * m1r + m1i * m1i);
      }
      __syncthreads();
      if (tid < BB) {
        float pp = probP[tid], pm = probM[tid];
        xcur[tid * XDIM + q] = pp - pm;          // snapshot value s
        scaleS[tid] = rsqrtf(pp);
        probP[tid] = 0.f; probM[tid] = 0.f;
      }
      __syncthreads();
      for (int i = tid; i < BB * DD; i += 256) {
        float sc = scaleS[i / DD];
        psiR[i] *= sc; psiI[i] *= sc;
      }
      __syncthreads();
    }

    // ---- record inputs[s] ----
    for (int i = tid; i < BB * XDIM; i += 256)
      inputs[s * BB * XDIM + i] = xcur[i];
    __syncthreads();
  }

  // ---- last_bv output (amps of last step) ----
  for (int i = tid; i < BB * NQ * KC; i += 256) {
    int b = i / (NQ * KC), j = i % (NQ * KC);
    lastbv_out[i] = xcur[b * XDIM + NQ + j];
  }
}

// ---------------- k_prep2: per-(b,s,q) 2x2 complex rho factors ----------------
__global__ void k_prep2(const float* __restrict__ inputs, float* __restrict__ rhoq) {
  int t = blockIdx.x * blockDim.x + threadIdx.x;
  if (t >= SS * BB * NQ) return;
  int q = t % NQ;
  int b = (t / NQ) % BB;
  int s = t / (NQ * BB);
  const float* x = inputs + (s * BB + b) * XDIM;
  float sh  = x[q];
  float bv0 = x[NQ + q * 3 + 0];
  float bv1 = x[NQ + q * 3 + 1];
  float bv2 = x[NQ + q * 3 + 2];
  float cc = 1.5f * sh;
  float* o = rhoq + (((size_t)b * SS + s) * NQ + q) * 8;
  o[0] = 0.5f + cc * bv2; o[1] = 0.f;          // m00
  o[2] = cc * bv0;        o[3] = -cc * bv1;    // m01 = 1.5 s (bv0 - i bv1)
  o[4] = cc * bv0;        o[5] = cc * bv1;     // m10
  o[6] = 0.5f - cc * bv2; o[7] = 0.f;          // m11
}

// ---------------- k_phase2: Kronecker reconstruction into d_out ----------------
__global__ void __launch_bounds__(256) k_phase2(const float* __restrict__ rhoq,
                                                float* __restrict__ out) {
  __shared__ float rq[SS * NQ * 8];            // 4.5 KB per block
  int b = blockIdx.x >> NQ;                    // / 512
  int row = blockIdx.x & (DD - 1);
  for (int i = threadIdx.x; i < SS * NQ * 8; i += 256)
    rq[i] = rhoq[(size_t)b * SS * NQ * 8 + i];
  __syncthreads();

  int rsel[NQ];
  for (int q = 0; q < NQ; ++q) rsel[q] = ((row >> (NQ - 1 - q)) & 1) * 2;

  for (int col = threadIdx.x; col < DD; col += 256) {
    float accR = 0.f, accI = 0.f;
    for (int s = 0; s < SS; ++s) {
      const float* base = rq + s * NQ * 8;
      float pr = 1.f, pi = 0.f;
      for (int q = 0; q < NQ; ++q) {
        int cb = (col >> (NQ - 1 - q)) & 1;
        const float* m = base + q * 8 + (rsel[q] + cb) * 2;
        float fr = m[0], fi = m[1];
        float nr = pr * fr - pi * fi;
        float ni = pr * fi + pi * fr;
        pr = nr; pi = ni;
      }
      accR += pr; accI += pi;
    }
    const float invS = 1.f / (float)SS;
    out[((size_t)(b * 2 + 0) * DD + row) * DD + col] = accR * invS;
    out[((size_t)(b * 2 + 1) * DD + row) * DD + col] = accI * invS;
  }
}

// ---------------- host launcher ----------------
extern "C" void kernel_launch(void* const* d_in, const int* in_sizes, int n_in,
                              void* d_out, int out_size, void* d_ws, size_t ws_size,
                              hipStream_t stream) {
  const float* snapshot = (const float*)d_in[0];
  const float* bcv      = (const float*)d_in[1];
  const float* rho      = (const float*)d_in[2];
  const float* h0       = (const float*)d_in[3];
  const float* c0       = (const float*)d_in[4];
  const float* W_ih     = (const float*)d_in[5];
  const float* W_hh     = (const float*)d_in[6];
  const float* b_ih     = (const float*)d_in[7];
  const float* b_hh     = (const float*)d_in[8];
  const float* Wp       = (const float*)d_in[9];
  const float* bp       = (const float*)d_in[10];

  char* ws = (char*)d_ws;
  _Float16* Wcat  = (_Float16*)(ws + WCAT_OFF);
  float*    biasS = (float*)   (ws + BIAS_OFF);
  _Float16* Bproj = (_Float16*)(ws + BPROJ_OFF);
  float*    bpPad = (float*)   (ws + BPPAD_OFF);
  float*    inputs= (float*)   (ws + INPUTS_OFF);
  float*    rhoq  = (float*)   (ws + RHOQ_OFF);

  float* out    = (float*)d_out;
  float* lastbv = out + (size_t)BB * 2 * DD * DD;   // tuple tail: last_bv [16][9][3]

  k_init<<<(512 * KU + 255) / 256, 256, 0, stream>>>(
      W_ih, W_hh, b_ih, b_hh, Wp, bp, Wcat, biasS, Bproj, bpPad);

  // dynamic LDS for the persistent phase-1 workgroup (~131 KB; CDNA5 WGP has 320 KB)
  constexpr int SMEM =
      (BB * KU + BB * HH) * 2 +                           // uf16 + hf16
      (BB * GN + 2 * BB * HH + 2 * BB * DD) * 4 +          // g, h, c, psiR, psiI
      (BB * XDIM + BB * NQ * KC + BB * 32 + 3 * BB) * 4;   // xcur, amps, logits, reductions
  (void)hipFuncSetAttribute(reinterpret_cast<const void*>(k_phase1),
                            hipFuncAttributeMaxDynamicSharedMemorySize, SMEM);
  k_phase1<<<1, 256, SMEM, stream>>>(snapshot, bcv, rho, h0, c0,
                                     Wcat, biasS, Bproj, bpPad, inputs, lastbv);

  k_prep2<<<(SS * BB * NQ + 255) / 256, 256, 0, stream>>>(inputs, rhoq);

  k_phase2<<<BB * DD, 256, 0, stream>>>(rhoq, out);
}
